// HybridNATModel_65481071404118
// MI455X (gfx1250) — matrix operand order repaired and checked by reading, not compiled, not run on hardware
//
#include <hip/hip_runtime.h>
#include <hip/hip_bf16.h>
#include <math.h>

// CDNA5 / gfx1250, wave32.
typedef __attribute__((ext_vector_type(16))) __bf16 v16bf;
typedef __attribute__((ext_vector_type(8)))  __bf16 v8bf;
typedef __attribute__((ext_vector_type(8)))  float  v8f;
typedef __attribute__((ext_vector_type(4)))  float  v4f;
typedef __attribute__((ext_vector_type(2)))  float  v2f;

#define BN_EPS 1e-5f

__device__ __forceinline__ __bf16 tobf(float f) { return (__bf16)f; }

__device__ __forceinline__ v16bf cat16(v8bf lo, v8bf hi) {
    return __builtin_shufflevector(lo, hi, 0,1,2,3,4,5,6,7,8,9,10,11,12,13,14,15);
}

__device__ __forceinline__ void wait_async0() {
#if __has_builtin(__builtin_amdgcn_s_wait_asynccnt)
    __builtin_amdgcn_s_wait_asynccnt(0);
#else
    asm volatile("s_wait_asynccnt 0x0" ::: "memory");
#endif
}

// Fused conv1+pool -> conv2+pool (WMMA bf16, pool folded as quadrant-GEMM max)
// -> em_w0 (WMMA bf16) -> 49x 4x4 relu chain -> fc1 (WMMA f32 16x16x4) -> fc2.
// 16 images per workgroup (2 per wave), 8 waves.
__global__ __launch_bounds__(256)
void hybrid_fused_kernel(const float* __restrict__ x,
                         const float* __restrict__ conv1_w, const float* __restrict__ conv1_b,
                         const float* __restrict__ conv2_w, const float* __restrict__ conv2_b,
                         const float* __restrict__ em_w0,   const float* __restrict__ em_ws,
                         const float* __restrict__ fc1_w,   const float* __restrict__ fc1_b,
                         const float* __restrict__ fc2_w,   const float* __restrict__ fc2_b,
                         float* __restrict__ prebn)
{
    __shared__ __align__(16) float  xpad[8][900];   // per-wave zero-padded input 30x30
    __shared__ __align__(16) __bf16 p1c[8][2048];   // per-wave pool1, [16y][16x][8ch], zero border
    __shared__ __align__(16) __bf16 flatb[16][800]; // per-image flat features (784 pad 800)
    __shared__ float  qbuf[16][4];
    __shared__ __align__(16) float  hbuf[16][64];
    __shared__ float  w1s[80];                      // conv1 w(72)+b(8)
    __shared__ __align__(16) __bf16 zb[8];          // zero tile for padded-K taps

    const int tid  = threadIdx.x;
    const int lane = tid & 31;
    const int wid  = tid >> 5;
    const int img0 = blockIdx.x * 16;

    if (tid < 72) w1s[tid]      = conv1_w[tid];
    if (tid < 8)  w1s[72 + tid] = conv1_b[tid];
    if (tid < 8)  zb[tid]       = tobf(0.f);
    __syncthreads();

    // ISA 7.12.2 lane mappings (wave32)
    const int pn   = lane & 15;
    const int hiA  = (lane & 16) ? 8  : 0;   // A: upper half-lanes hold K+8 in each 16-group
    const int hiB  = (lane & 16) ? 16 : 0;   // A and B here: upper half-lanes hold K+16 (contiguous k order)
    const int mrow = (lane & 16) ? 8  : 0;   // C/D: row = vgpr + mrow

    // conv2 A fragments, channel-last K ordering: K = tap*8 + cc (tap = dy*3+dx), pad 72->96
    v16bf aW[3];
#pragma unroll
    for (int ks = 0; ks < 3; ++ks) {
#pragma unroll
        for (int e = 0; e < 16; ++e) {
            int K   = ks * 32 + hiB + e;
            int tap = K >> 3, cc = K & 7;
            float v = (tap < 9) ? conv2_w[pn * 72 + cc * 9 + tap] : 0.f;
            aW[ks][e] = tobf(v);
        }
    }

    // per-lane B-operand tap geometry: k-step ks needs taps t = ks*4 + (hi?2:0) + {0,1}
    const int tb = (lane & 16) ? 2 : 0;
    int  toff[3][2];
    bool tok[3][2];
#pragma unroll
    for (int ks = 0; ks < 3; ++ks)
#pragma unroll
        for (int j = 0; j < 2; ++j) {
            int t = ks * 4 + tb + j;
            int dy = t / 3, dx = t - 3 * dy;
            toff[ks][j] = dy * 16 + dx;
            tok[ks][j]  = (t < 9);
        }

    for (int s = 0; s < 2; ++s) {
        const int iml = wid * 2 + s;
        const float* xim = x + (size_t)(img0 + iml) * 784;

        // zero pool1 buffer (vector stores) and xpad *border only* (keeps DS writes
        // disjoint from the async interior fill -> no DS/ASYNC ordering hazard)
        v8bf z8 = *(const v8bf*)zb;
#pragma unroll
        for (int j = 0; j < 8; ++j)
            *(v8bf*)&p1c[wid][(lane + 32 * j) * 8] = z8;
#pragma unroll
        for (int j = 0; j < 4; ++j) {
            int idx = lane + 32 * j;
            if (idx < 116) {
                int r, c;
                if      (idx < 30) { r = 0;  c = idx; }
                else if (idx < 60) { r = 29; c = idx - 30; }
                else { int rem = idx - 60; r = 1 + (rem >> 1); c = (rem & 1) ? 29 : 0; }
                xpad[wid][r * 30 + c] = 0.f;
            }
        }

        // async-copy image into padded interior: per-lane scatter, GVS addressing
        for (int j = 0; j < 25; ++j) {
            int i = lane + 32 * j;
            if (i < 784) {
                int y = i / 28, xx = i - 28 * y;
                unsigned ldsa = (unsigned)(size_t)&xpad[wid][(y + 1) * 30 + (xx + 1)];
                asm volatile("global_load_async_to_lds_b32 %0, %1, %2"
                             :: "v"(ldsa), "v"(i * 4), "s"(xim) : "memory");
            }
        }
        wait_async0();

        // ---- conv1 + relu + maxpool2 -> p1c (branchless, shared 4x4 window) ----
        for (int i = lane; i < 1568; i += 32) {
            int ch  = i / 196;
            int rem = i - ch * 196;
            int py  = rem / 14;
            int px  = rem - 14 * py;
            float win[4][4];
#pragma unroll
            for (int r = 0; r < 4; ++r)
#pragma unroll
                for (int c = 0; c < 4; ++c)
                    win[r][c] = xpad[wid][(py * 2 + r) * 30 + (px * 2 + c)];
            float m = -1e30f;
#pragma unroll
            for (int q = 0; q < 4; ++q) {
                int qy = q >> 1, qx = q & 1;
                float acc = w1s[72 + ch];
#pragma unroll
                for (int dy = 0; dy < 3; ++dy)
#pragma unroll
                    for (int dx = 0; dx < 3; ++dx)
                        acc = fmaf(win[qy + dy][qx + dx], w1s[ch * 9 + dy * 3 + dx], acc);
                m = fmaxf(m, acc);
            }
            p1c[wid][((py + 1) * 16 + (px + 1)) * 8 + ch] = tobf(fmaxf(m, 0.f));
        }
        for (int i = 784 + lane; i < 800; i += 32) flatb[iml][i] = tobf(0.f);

        // ---- conv2 via WMMA bf16; 2x ds_load_b128 per B fragment ----
#pragma unroll
        for (int tile = 0; tile < 4; ++tile) {
            int pp   = tile * 16 + pn;
            bool ppok = (pp < 49);
            int ppc  = ppok ? pp : 0;
            int ppy  = ppc / 7, ppx = ppc - 7 * ppy;
            v8f dmax;
#pragma unroll
            for (int q = 0; q < 4; ++q) {
                int base = (ppy * 2 + (q >> 1)) * 16 + (ppx * 2 + (q & 1));
                v8f c = {0.f, 0.f, 0.f, 0.f, 0.f, 0.f, 0.f, 0.f};
#pragma unroll
                for (int ks = 0; ks < 3; ++ks) {
                    const __bf16* s0 = tok[ks][0] ? &p1c[wid][(base + toff[ks][0]) * 8] : zb;
                    const __bf16* s1 = tok[ks][1] ? &p1c[wid][(base + toff[ks][1]) * 8] : zb;
                    v16bf b = cat16(*(const v8bf*)s0, *(const v8bf*)s1);
                    c = __builtin_amdgcn_wmma_f32_16x16x32_bf16(
                            false, aW[ks], false, b, (short)0, c, false, false);
                }
                if (q == 0) dmax = c;
                else {
#pragma unroll
                    for (int g = 0; g < 8; ++g) dmax[g] = fmaxf(dmax[g], c[g]);
                }
            }
#pragma unroll
            for (int g = 0; g < 8; ++g) {
                int ch = g + mrow;
                float v = fmaxf(dmax[g] + conv2_b[ch], 0.f);
                if (ppok) flatb[iml][ch * 49 + pp] = tobf(v);
            }
        }
    }
    __syncthreads();

    // ---- q = relu(flat[16x784] @ em_w0^T) via WMMA bf16, wave 0 ----
    if (wid == 0) {
        const float* wp = em_w0 + (size_t)(pn & 3) * 784;  // clamp -> branchless, no OOB
        v8f c = {0.f, 0.f, 0.f, 0.f, 0.f, 0.f, 0.f, 0.f};
        for (int ks = 0; ks < 24; ++ks) {
            v16bf a = cat16(*(const v8bf*)&flatb[pn][ks * 32 + hiB],
                            *(const v8bf*)&flatb[pn][ks * 32 + hiB + 8]);
            const float* bp = wp + ks * 32 + hiB;
            v16bf b;
#pragma unroll
            for (int e4 = 0; e4 < 4; ++e4) {
                v4f f4 = *(const v4f*)(bp + e4 * 4);
#pragma unroll
                for (int u = 0; u < 4; ++u) b[e4 * 4 + u] = tobf(f4[u]);
            }
            c = __builtin_amdgcn_wmma_f32_16x16x32_bf16(
                    false, a, false, b, (short)0, c, false, false);
        }
        {   // tail k-step: K 768..799; B valid only for lower half-lanes (768..783)
            v16bf a = cat16(*(const v8bf*)&flatb[pn][768 + hiB],
                            *(const v8bf*)&flatb[pn][768 + hiB + 8]);
            v16bf b;
            if (hiB == 0) {
                const float* bp = wp + 768;
#pragma unroll
                for (int e4 = 0; e4 < 4; ++e4) {
                    v4f f4 = *(const v4f*)(bp + e4 * 4);
#pragma unroll
                    for (int u = 0; u < 4; ++u) b[e4 * 4 + u] = tobf(f4[u]);
                }
            } else {
#pragma unroll
                for (int e = 0; e < 16; ++e) b[e] = tobf(0.f);
            }
            c = __builtin_amdgcn_wmma_f32_16x16x32_bf16(
                    false, a, false, b, (short)0, c, false, false);
        }
#pragma unroll
        for (int g = 0; g < 8; ++g) {
            int im = g + mrow;
            if (pn < 4) qbuf[im][pn] = fmaxf(c[g], 0.f);
        }

        // ---- 49 sequential 4x4 relu layers (single wave, LDS in-order) ----
        for (int layer = 0; layer < 49; ++layer) {
            const float* w = em_ws + layer * 16;
            float nv[2];
#pragma unroll
            for (int t = 0; t < 2; ++t) {
                int p = lane + t * 32;
                int im = p >> 2, f = p & 3;
                float acc = 0.f;
#pragma unroll
                for (int k = 0; k < 4; ++k) acc = fmaf(qbuf[im][k], w[f * 4 + k], acc);
                nv[t] = fmaxf(acc, 0.f);
            }
#pragma unroll
            for (int t = 0; t < 2; ++t) {
                int p = lane + t * 32;
                qbuf[p >> 2][p & 3] = nv[t];
            }
        }
    }
    __syncthreads();

    // ---- fc1: [16x4]@[4x64] via v_wmma_f32_16x16x4_f32, one 16-col tile per wave ----
    if (wid < 4) {
        int koff = (lane & 16) ? 2 : 0;
        v2f a, b;
        a[0] = qbuf[pn][koff];
        a[1] = qbuf[pn][koff + 1];
        int n = wid * 16 + pn;
        b[0] = fc1_w[n * 4 + koff];
        b[1] = fc1_w[n * 4 + koff + 1];
        v8f c0 = {0.f, 0.f, 0.f, 0.f, 0.f, 0.f, 0.f, 0.f};
        v8f d = __builtin_amdgcn_wmma_f32_16x16x4_f32(
                    false, a, false, b, (short)0, c0, false, false);
#pragma unroll
        for (int g = 0; g < 8; ++g)
            hbuf[g + mrow][n] = fmaxf(d[g] + fc1_b[n], 0.f);
    }
    __syncthreads();

    // ---- fc2 (64->4), vectorized LDS reads, write pre-BN output ----
    if (wid == 0) {
#pragma unroll
        for (int t = 0; t < 2; ++t) {
            int p = lane + t * 32;
            int im = p >> 2, f = p & 3;
            const float* w2 = fc2_w + f * 64;
            float acc = fc2_b[f];
#pragma unroll
            for (int k4 = 0; k4 < 16; ++k4) {
                v4f h4 = *(const v4f*)&hbuf[im][k4 * 4];
                v4f w4 = *(const v4f*)&w2[k4 * 4];
                acc = fmaf(h4[0], w4[0], acc);
                acc = fmaf(h4[1], w4[1], acc);
                acc = fmaf(h4[2], w4[2], acc);
                acc = fmaf(h4[3], w4[3], acc);
            }
            prebn[(size_t)(img0 + im) * 4 + f] = acc;
        }
    }
}

// Deterministic batch statistics: one block per feature, tree reduction.
__global__ __launch_bounds__(256)
void bn_stats_kernel(const float* __restrict__ prebn, float* __restrict__ ws, int B)
{
    __shared__ float ss[256], sq[256];
    const int f = blockIdx.x;
    float s = 0.f, s2 = 0.f;
    for (int i = threadIdx.x; i < B; i += 256) {
        float v = prebn[(size_t)i * 4 + f];
        s += v; s2 += v * v;
    }
    ss[threadIdx.x] = s; sq[threadIdx.x] = s2;
    __syncthreads();
    for (int st = 128; st > 0; st >>= 1) {
        if (threadIdx.x < st) {
            ss[threadIdx.x] += ss[threadIdx.x + st];
            sq[threadIdx.x] += sq[threadIdx.x + st];
        }
        __syncthreads();
    }
    if (threadIdx.x == 0) {
        float mean = ss[0] / (float)B;
        float var  = sq[0] / (float)B - mean * mean;
        ws[f]     = mean;
        ws[4 + f] = rsqrtf(var + BN_EPS);
    }
}

__global__ __launch_bounds__(256)
void bn_apply_kernel(float* __restrict__ out, const float* __restrict__ ws,
                     const float* __restrict__ gamma, const float* __restrict__ beta, int total)
{
    int i = blockIdx.x * blockDim.x + threadIdx.x;
    if (i < total) {
        int f = i & 3;
        out[i] = (out[i] - ws[f]) * ws[4 + f] * gamma[f] + beta[f];
    }
}

extern "C" void kernel_launch(void* const* d_in, const int* in_sizes, int n_in,
                              void* d_out, int out_size, void* d_ws, size_t ws_size,
                              hipStream_t stream)
{
    const float* x       = (const float*)d_in[0];
    const float* conv1_w = (const float*)d_in[1];
    const float* conv1_b = (const float*)d_in[2];
    const float* conv2_w = (const float*)d_in[3];
    const float* conv2_b = (const float*)d_in[4];
    const float* em_w0   = (const float*)d_in[5];
    const float* em_ws   = (const float*)d_in[6];
    const float* fc1_w   = (const float*)d_in[7];
    const float* fc1_b   = (const float*)d_in[8];
    const float* fc2_w   = (const float*)d_in[9];
    const float* fc2_b   = (const float*)d_in[10];
    const float* bn_g    = (const float*)d_in[11];
    const float* bn_b    = (const float*)d_in[12];

    float* out = (float*)d_out;
    float* ws  = (float*)d_ws;   // 8 floats: mean[4], rstd[4]

    const int B = in_sizes[0] / 784;      // 16384
    const int nblocks = B / 16;           // 16 images per workgroup

    hybrid_fused_kernel<<<nblocks, 256, 0, stream>>>(
        x, conv1_w, conv1_b, conv2_w, conv2_b, em_w0, em_ws,
        fc1_w, fc1_b, fc2_w, fc2_b, out);
    bn_stats_kernel<<<4, 256, 0, stream>>>(out, ws, B);
    bn_apply_kernel<<<(B * 4 + 255) / 256, 256, 0, stream>>>(out, ws, bn_g, bn_b, B * 4);
}
